// MultiHeadAttention_17111149707675
// MI455X (gfx1250) — matrix operand order repaired and checked by reading, not compiled, not run on hardware
//
#include <hip/hip_runtime.h>
#include <hip/hip_bf16.h>

// ---------------------------------------------------------------------------
// MHA forward for gfx1250 (wave32, WMMA 16x16x32 f16, f32 accumulate)
// Pipeline:
//   1) wconv:     f32 weights -> f16 (optionally pre-scaled), transposed [N][K]
//   2) maskpack:  bool mask [B,S,S] -> bit-packed u32 words [B,S,S/32]
//   3) proj x3:   one wave = 16 seq rows x 4 heads (A-fragment reuse);
//                 Q,K -> [B,H,S,32] f16, V -> [B,H,32,S] f16 (transposed store)
//   4) flash:     4 waves/block share (b,h) -> K/V loads hit WGP$; each wave
//                 owns a 16-query tile, streams 32-key blocks with online
//                 softmax (LDS C->A remap for the P tile)
//   5) out GEMM:  one wave = 16 rows x 128 cols (A-fragment reuse), f32 out
// ---------------------------------------------------------------------------

constexpr int kB  = 4;
constexpr int kS  = 2048;
constexpr int kD  = 1024;
constexpr int kH  = 16;
constexpr int kHD = 32;
constexpr float kInvScale = 0.125f;   // 1/sqrt(DEPTH=64), folded into Wq
constexpr float kNeg = -1e10f;

typedef __attribute__((ext_vector_type(16))) _Float16 v16h;
typedef __attribute__((ext_vector_type(8)))  _Float16 v8h;
typedef __attribute__((ext_vector_type(8)))  float    v8f;

union V16U { v16h v; v8h h[2]; };

__device__ __forceinline__ v8f wmma_f16(v16h a, v16h b, v8f c) {
  // (neg_a, A, neg_b, B, c_mod, C, reuse_a, reuse_b)
  return __builtin_amdgcn_wmma_f32_16x16x32_f16(false, a, false, b, (short)0, c,
                                                false, false);
}

// 16-bit A fragment (16x32, M=lane%16): two contiguous 8-half runs per lane.
// lanes 0-15: K runs [0,8) and [16,24); lanes 16-31: [8,16) and [24,32).
__device__ __forceinline__ v16h load_a16(const _Float16* row, bool lo) {
  V16U u;
  u.h[0] = *(const v8h*)(row + (lo ? 0 : 8));
  u.h[1] = *(const v8h*)(row + (lo ? 16 : 24));
  return u.v;
}

__device__ __forceinline__ v16h load_a16_f32(const float* row, bool lo) {
  v16h r;
  const int k0 = lo ? 0 : 8;
  const int k1 = lo ? 16 : 24;
#pragma unroll
  for (int i = 0; i < 8; ++i) r[i] = (_Float16)row[k0 + i];
#pragma unroll
  for (int i = 0; i < 8; ++i) r[8 + i] = (_Float16)row[k1 + i];
  return r;
}

// 16-bit B fragment (32x16, N=lane%16): two contiguous 8-half runs per lane.
// lanes 0-15: K runs [0,8) and [16,24); lanes 16-31: [8,16) and [24,32). `col` points at K=0 of column N.
__device__ __forceinline__ v16h load_b16(const _Float16* col, bool lo) {
  V16U u;
  u.h[0] = *(const v8h*)(col + (lo ? 0 : 8));
  u.h[1] = *(const v8h*)(col + (lo ? 16 : 24));
  return u.v;
}

// ---------------------------------------------------------------------------
// Weight convert+transpose: in [HH][DD][EE] f32 -> out [HH][EE][DD] f16 * scale
// ---------------------------------------------------------------------------
__global__ void mha_wconv_kernel(const float* __restrict__ in,
                                 _Float16* __restrict__ out,
                                 int HH, int DD, int EE, float scale) {
  size_t i = (size_t)blockIdx.x * blockDim.x + threadIdx.x;
  size_t total = (size_t)HH * DD * EE;
  if (i >= total) return;
  int e = (int)(i % EE);
  size_t t = i / EE;
  int d = (int)(t % DD);
  int hh = (int)(t / DD);
  out[((size_t)hh * EE + e) * DD + d] = (_Float16)(in[i] * scale);
}

// ---------------------------------------------------------------------------
// Bool mask -> bit-packed words: out[b*S + q][w] bit j = mask[b,q,w*32+j]
// ---------------------------------------------------------------------------
__global__ void mha_maskpack_kernel(const unsigned char* __restrict__ m,
                                    unsigned* __restrict__ out, size_t nwords) {
  size_t i = (size_t)blockIdx.x * blockDim.x + threadIdx.x;
  if (i >= nwords) return;
  const uint4* p = (const uint4*)(m + i * 32);
  uint4 a = p[0], bq = p[1];
  unsigned vals[8] = {a.x, a.y, a.z, a.w, bq.x, bq.y, bq.z, bq.w};
  unsigned w = 0;
#pragma unroll
  for (int j = 0; j < 8; ++j) {
    unsigned v = vals[j];
    w |= ((v & 0x000000FFu) ? 1u : 0u) << (j * 4 + 0);
    w |= ((v & 0x0000FF00u) ? 1u : 0u) << (j * 4 + 1);
    w |= ((v & 0x00FF0000u) ? 1u : 0u) << (j * 4 + 2);
    w |= ((v & 0xFF000000u) ? 1u : 0u) << (j * 4 + 3);
  }
  out[i] = w;
}

// ---------------------------------------------------------------------------
// Per-head projection: one wave -> 16(M seq) x 4 heads x 32(N hd), K=1024.
// A-fragment (incl. f32->f16 convert) reused across 4 heads per k-step;
// the 4 waves of a block share the same 16 seq rows -> A loads hit WGP$.
// storeT=0: dst [B,H,S,HD] row-major (Q,K);  storeT=1: dst [B,H,HD,S] (V^T)
// ---------------------------------------------------------------------------
__global__ void __launch_bounds__(128) mha_proj_kernel(
    const float* __restrict__ src, const _Float16* __restrict__ Wt,
    _Float16* __restrict__ dst, int storeT) {
  const int lane = threadIdx.x & 31;
  const int wid  = threadIdx.x >> 5;
  const int hg   = wid;                      // head group of 4 (per wave)
  const int tile = blockIdx.x;               // 0..511 (shared by block)
  const int b    = tile >> 7;                // tile / (S/16)
  const int s0   = (tile & 127) << 4;
  const bool lo  = lane < 16;
  const int n    = lane & 15;
  const int h0   = hg * 4;

  const float* arow = src + ((size_t)b * kS + s0 + n) * kD;

  v8f acc[8];
#pragma unroll
  for (int i = 0; i < 8; ++i) acc[i] = (v8f){};

  for (int k = 0; k < kD; k += 32) {
    v16h a = load_a16_f32(arow + k, lo);
#pragma unroll
    for (int hh = 0; hh < 4; ++hh) {
      const _Float16* w = Wt + (size_t)(h0 + hh) * kHD * kD;
      v16h b0 = load_b16(w + (size_t)n * kD + k, lo);
      v16h b1 = load_b16(w + (size_t)(16 + n) * kD + k, lo);
      acc[2 * hh]     = wmma_f16(a, b0, acc[2 * hh]);
      acc[2 * hh + 1] = wmma_f16(a, b1, acc[2 * hh + 1]);
    }
  }

#pragma unroll
  for (int hh = 0; hh < 4; ++hh) {
    const size_t bh = (size_t)(b * kH + h0 + hh);
    if (!storeT) {
      // C layout: VGPR r -> row r (lanes 0-15) / r+8 (lanes 16-31), col=lane%16
#pragma unroll
      for (int r = 0; r < 8; ++r) {
        const int row = s0 + r + (lo ? 0 : 8);
        _Float16* o = dst + (bh * kS + row) * kHD;
        o[n]      = (_Float16)acc[2 * hh][r];
        o[16 + n] = (_Float16)acc[2 * hh + 1][r];
      }
    } else {
      // transposed store is contiguous along M: one b128 store per accumulator
      const int base = s0 + (lo ? 0 : 8);
      v8h o0, o1;
#pragma unroll
      for (int r = 0; r < 8; ++r) {
        o0[r] = (_Float16)acc[2 * hh][r];
        o1[r] = (_Float16)acc[2 * hh + 1][r];
      }
      *(v8h*)(dst + (bh * kHD + n) * kS + base)      = o0;
      *(v8h*)(dst + (bh * kHD + 16 + n) * kS + base) = o1;
    }
  }
}

// ---------------------------------------------------------------------------
// Flash attention. Block = 4 waves sharing (b,h), 4 adjacent 16-query tiles:
// all waves issue identical K/V fragment loads -> served by WGP$ after the
// first miss. Each wave streams 32-key blocks with online softmax.
// ---------------------------------------------------------------------------
__global__ void __launch_bounds__(128) mha_flash_kernel(
    const _Float16* __restrict__ Qh, const _Float16* __restrict__ Kh,
    const _Float16* __restrict__ Vt, const unsigned* __restrict__ maskw,
    _Float16* __restrict__ multi) {
  __shared__ __align__(16) _Float16 pb[4][16 * 32];  // per-wave P tile [16q][32k]
  const int lane = threadIdx.x & 31;
  const int wid  = threadIdx.x >> 5;
  const int bhi  = blockIdx.x >> 5;        // 0..63: (b,h) shared by block
  const int qg   = blockIdx.x & 31;        // query group of 4 tiles
  const int b    = bhi >> 4;
  const int h    = bhi & 15;
  const int q0   = ((qg << 2) + wid) << 4; // this wave's 16-query tile
  const bool lo  = lane < 16;
  const int n    = lane & 15;
  _Float16* pl = pb[wid];

  const size_t bh = (size_t)(b * kH + h);
  v16h qfrag = load_a16(Qh + (bh * kS + q0 + n) * kHD, lo);

  v8f accA = {}; v8f accB = {};
  float mrow[8], lrow[8];
#pragma unroll
  for (int r = 0; r < 8; ++r) { mrow[r] = -3.0e38f; lrow[r] = 0.0f; }

  // lane holds packed-mask words for query row q0 + (lane&15)
  const unsigned* mwrow = maskw + ((size_t)b * kS + q0 + n) * (kS / 32);
  const int srcLane = (lo ? 0 : 8);   // row r lives in lane r (lo) / r+8 (hi)

  for (int kb = 0; kb < kS; kb += 32) {
    // ---- scores: Q(16x32) x K^T -> two 16x16 f32 tiles (scale folded in Wq)
    v16h kf0 = load_b16(Kh + (bh * kS + kb + n) * kHD, lo);
    v16h kf1 = load_b16(Kh + (bh * kS + kb + 16 + n) * kHD, lo);
    // V fragments issued early: consumed only after the softmax VALU work
    v16h vf0 = load_b16(Vt + (bh * kHD + n) * kS + kb, lo);
    v16h vf1 = load_b16(Vt + (bh * kHD + 16 + n) * kS + kb, lo);
    // one packed word covers this lane's row for all 32 keys of the block
    const unsigned wl = mwrow[kb >> 5];

    v8f z = {};
    v8f s0 = wmma_f16(qfrag, kf0, z);
    v8f s1 = wmma_f16(qfrag, kf1, z);

    // prefetch K and V two key blocks ahead into cache (global_prefetch_b8)
    if (kb + 64 < kS) {
      __builtin_prefetch(Kh + (bh * kS + kb + 64 + n) * kHD, 0, 3);
      __builtin_prefetch(Vt + (bh * kHD + n) * kS + kb + 64, 0, 3);
    }

    // ---- online softmax per row (rows live across 16-lane halves)
#pragma unroll
    for (int r = 0; r < 8; ++r) {
      const int row = r + (lo ? 0 : 8);
      const unsigned wm = (unsigned)__shfl((int)wl, srcLane + r, 32);
      float a0 = s0[r];
      float a1 = s1[r];
      a0 = ((wm >> n) & 1u)        ? kNeg : a0;
      a1 = ((wm >> (16 + n)) & 1u) ? kNeg : a1;
      float mx = fmaxf(a0, a1);
#pragma unroll
      for (int off = 8; off >= 1; off >>= 1)
        mx = fmaxf(mx, __shfl_xor(mx, off, 32));
      const float mnew = fmaxf(mrow[r], mx);
      const float corr = __expf(mrow[r] - mnew);
      const float p0 = __expf(a0 - mnew);
      const float p1 = __expf(a1 - mnew);
      float rs = p0 + p1;
#pragma unroll
      for (int off = 8; off >= 1; off >>= 1)
        rs += __shfl_xor(rs, off, 32);
      lrow[r] = lrow[r] * corr + rs;
      mrow[r] = mnew;
      accA[r] *= corr;
      accB[r] *= corr;
      // C-layout -> LDS [row][key] for A-fragment remap
      pl[row * 32 + n]      = (_Float16)p0;
      pl[row * 32 + 16 + n] = (_Float16)p1;
    }

    // ---- z += P(16x32) x V(32keys x 32hd); V^T layout makes B-frags contiguous
    v16h pfrag = load_a16(pl + n * 32, lo);
    accA = wmma_f16(pfrag, vf0, accA);
    accB = wmma_f16(pfrag, vf1, accB);
  }

  // ---- normalize and write concatenated heads: multi [B,S,H*HD] f16
#pragma unroll
  for (int r = 0; r < 8; ++r) {
    const int qrow = q0 + r + (lo ? 0 : 8);
    const float inv = 1.0f / lrow[r];
    _Float16* o = multi + ((size_t)b * kS + qrow) * (kH * kHD) + h * kHD;
    o[n]      = (_Float16)(accA[r] * inv);
    o[16 + n] = (_Float16)(accB[r] * inv);
  }
}

// ---------------------------------------------------------------------------
// Output GEMM: multi [8192,512] f16 x Wo^T [1024,512] f16 -> out [8192,1024] f32
// One wave = 16 rows x 128 cols (4 col-tiles reuse the A fragment); the 4
// waves of a block share the same 16 rows -> A loads hit WGP$.
// ---------------------------------------------------------------------------
__global__ void __launch_bounds__(128) mha_out_kernel(
    const _Float16* __restrict__ multi, const _Float16* __restrict__ Wot,
    float* __restrict__ out) {
  const int lane = threadIdx.x & 31;
  const int wid  = threadIdx.x >> 5;
  const int cg   = (blockIdx.x & 1) * 4 + wid;  // col group: 128 cols
  const int rt   = blockIdx.x >> 1;             // 0..511
  const int r0   = rt << 4;
  const bool lo  = lane < 16;
  const int n    = lane & 15;
  constexpr int kIn = kH * kHD;                 // 512

  const _Float16* arow = multi + (size_t)(r0 + n) * kIn;

  v8f acc[8];
#pragma unroll
  for (int i = 0; i < 8; ++i) acc[i] = (v8f){};

  for (int k = 0; k < kIn; k += 32) {
    v16h a = load_a16(arow + k, lo);
#pragma unroll
    for (int cc = 0; cc < 4; ++cc) {
      const int cbase = cg * 128 + cc * 32;
      v16h b0 = load_b16(Wot + (size_t)(cbase + n) * kIn + k, lo);
      v16h b1 = load_b16(Wot + (size_t)(cbase + 16 + n) * kIn + k, lo);
      acc[2 * cc]     = wmma_f16(a, b0, acc[2 * cc]);
      acc[2 * cc + 1] = wmma_f16(a, b1, acc[2 * cc + 1]);
    }
  }

#pragma unroll
  for (int cc = 0; cc < 4; ++cc) {
    const int cbase = cg * 128 + cc * 32;
#pragma unroll
    for (int r = 0; r < 8; ++r) {
      const int row = r0 + r + (lo ? 0 : 8);
      float* o = out + (size_t)row * kD + cbase;
      o[n]      = acc[2 * cc][r];
      o[16 + n] = acc[2 * cc + 1][r];
    }
  }
}

// ---------------------------------------------------------------------------
extern "C" void kernel_launch(void* const* d_in, const int* in_sizes, int n_in,
                              void* d_out, int out_size, void* d_ws, size_t ws_size,
                              hipStream_t stream) {
  const float* q  = (const float*)d_in[0];
  const float* k  = (const float*)d_in[1];
  const float* v  = (const float*)d_in[2];
  const unsigned char* mask = (const unsigned char*)d_in[3];
  const float* Wq = (const float*)d_in[4];
  const float* Wk = (const float*)d_in[5];
  const float* Wv = (const float*)d_in[6];
  const float* Wo = (const float*)d_in[7];

  constexpr size_t szQ  = (size_t)kB * kH * kS * kHD;  // 4,194,304 f16 elems each
  constexpr size_t szW  = (size_t)kH * kHD * kD;       // 524,288
  constexpr size_t szWo = (size_t)kD * (kH * kHD);     // 524,288
  constexpr size_t szMW = (size_t)kB * kS * (kS / 32); // 524,288 u32 words

  _Float16* ws    = (_Float16*)d_ws;
  _Float16* Qh    = ws; ws += szQ;   // [B,H,S,HD]
  _Float16* Kh    = ws; ws += szQ;   // [B,H,S,HD]
  _Float16* Vt    = ws; ws += szQ;   // [B,H,HD,S]
  _Float16* multi = ws; ws += szQ;   // [B,S,H*HD]
  _Float16* Wqt   = ws; ws += szW;   // [H,HD,D]
  _Float16* Wkt   = ws; ws += szW;
  _Float16* Wvt   = ws; ws += szW;
  _Float16* Wot   = ws; ws += szWo;  // [D, H*HD]
  unsigned* maskw = (unsigned*)ws;   // [B,S,S/32]

  const int blk = 256;
  mha_wconv_kernel<<<(unsigned)((szW + blk - 1) / blk), blk, 0, stream>>>(
      Wq, Wqt, kH, kD, kHD, kInvScale);   // fold 1/sqrt(depth) into Wq
  mha_wconv_kernel<<<(unsigned)((szW + blk - 1) / blk), blk, 0, stream>>>(
      Wk, Wkt, kH, kD, kHD, 1.0f);
  mha_wconv_kernel<<<(unsigned)((szW + blk - 1) / blk), blk, 0, stream>>>(
      Wv, Wvt, kH, kD, kHD, 1.0f);
  mha_wconv_kernel<<<(unsigned)((szWo + blk - 1) / blk), blk, 0, stream>>>(
      Wo, Wot, 1, kH * kHD, kD, 1.0f);
  mha_maskpack_kernel<<<(unsigned)((szMW + blk - 1) / blk), blk, 0, stream>>>(
      mask, maskw, szMW);

  mha_proj_kernel<<<512, 128, 0, stream>>>(q, Wqt, Qh, 0);
  mha_proj_kernel<<<512, 128, 0, stream>>>(k, Wkt, Kh, 0);
  mha_proj_kernel<<<512, 128, 0, stream>>>(v, Wvt, Vt, 1);

  mha_flash_kernel<<<2048, 128, 0, stream>>>(Qh, Kh, Vt, maskw, multi);

  mha_out_kernel<<<1024, 128, 0, stream>>>(multi, Wot, (float*)d_out);
}
